// LXRTXLayer_90297392431751
// MI455X (gfx1250) — matrix operand order repaired and checked by reading, not compiled, not run on hardware
//
#include <hip/hip_runtime.h>

// ---------------------------------------------------------------------------
// CDNA5 (gfx1250) LXMERT cross-modal layer.
// All matmuls via v_wmma_f32_16x16x32_bf16 (bf16 in, f32 accumulate).
// wave32; 8 waves / 256-thread block; 2x2 register tiling per wave
// (4 independent WMMA accumulator chains -> hides WMMA RAW hazard, halves
// redundant fragment traffic vs 1x2 tiling).
// ---------------------------------------------------------------------------

typedef __attribute__((ext_vector_type(16))) __bf16 v16bf;
typedef __attribute__((ext_vector_type(8)))  float  v8f;

#define WMMA_BF16(a, b, c) \
  __builtin_amdgcn_wmma_f32_16x16x32_bf16(false, (a), false, (b), (short)0, (c), false, false)

__device__ __forceinline__ unsigned short f2bf(float f) {
  unsigned int x = __float_as_uint(f);
  x += 0x7FFFu + ((x >> 16) & 1u);          // round-to-nearest-even
  return (unsigned short)(x >> 16);
}

// A fragment: 16x32 bf16, row-major source, ld in elements (even).
// ISA layout: lanes 0-15 -> M=lane, K pairs {0..7,16..23}; lanes 16-31 -> K+8.
__device__ __forceinline__ v16bf load_a_frag(const unsigned short* A, int lda,
                                             int row0, int kb, int lane) {
  int m = row0 + (lane & 15);
  int kbase = kb + ((lane >> 4) << 3);
  const unsigned int* p = (const unsigned int*)(A + (size_t)m * lda + kbase);
  union { v16bf v; unsigned int u[8]; } r;
  r.u[0] = p[0]; r.u[1] = p[1]; r.u[2] = p[2];  r.u[3] = p[3];
  r.u[4] = p[8]; r.u[5] = p[9]; r.u[6] = p[10]; r.u[7] = p[11];
  return r.v;
}

// B fragment: 32x16 bf16 from a [N,K] row-major source (pre-transposed weight,
// K rows of the key matrix, or d-major V^T). lane = N column; lanes 0-15 hold
// K={0..7,16..23}, lanes 16-31 hold K+8 (same order as the A fragment).
__device__ __forceinline__ v16bf load_bt_frag(const unsigned short* Bt, int ldb,
                                              int col0, int kb, int lane) {
  int n = col0 + (lane & 15);
  int kbase = kb + ((lane >> 4) << 3);
  const unsigned int* p = (const unsigned int*)(Bt + (size_t)n * ldb + kbase);
  union { v16bf v; unsigned int u[8]; } r;
#pragma unroll
  for (int j = 0; j < 8; ++j) r.u[j] = p[j + ((j >> 2) << 2)];
  return r.v;
}

// C/D 16x16 f32 tile store: lane n = lane&15; VGPR r -> M = r + (lane>>4)*8.
__device__ __forceinline__ void store_tile_f32(float* C, int ldc, int m0, int n0,
                                               int lane, const v8f& c, float bv) {
  int n = lane & 15, mb = (lane >> 4) << 3;
#pragma unroll
  for (int r = 0; r < 8; ++r)
    C[(size_t)(m0 + mb + r) * ldc + n0 + n] = c[r] + bv;
}

__device__ __forceinline__ void store_tile_bf16(unsigned short* C, int ldc, int m0, int n0,
                                                int lane, const v8f& c, float bv) {
  int n = lane & 15, mb = (lane >> 4) << 3;
#pragma unroll
  for (int r = 0; r < 8; ++r)
    C[(size_t)(m0 + mb + r) * ldc + n0 + n] = f2bf(c[r] + bv);
}

// ---------------------------------------------------------------------------
// Generic GEMM: C[M,N] = A[M,K](bf16) * Bt[N,K](bf16)^T + bias, out f32/bf16.
// Block tile 128x64 (8 waves, 4Mx2N), wave tile 32x32 (2x2 WMMA tiles).
// ---------------------------------------------------------------------------
__global__ void k_gemm(const unsigned short* __restrict__ A, int lda,
                       const unsigned short* __restrict__ Bt, int ldb,
                       float* __restrict__ Cf, unsigned short* __restrict__ Cb,
                       int ldc, int M, int N, int K,
                       const float* __restrict__ bias) {
  int lane = threadIdx.x & 31;
  int wave = threadIdx.x >> 5;
  int m0 = blockIdx.y * 128 + (wave & 3) * 32;
  int n0 = blockIdx.x * 64 + (wave >> 2) * 32;
  if (m0 >= M || n0 >= N) return;               // wave-uniform guard
  v8f c00 = {0, 0, 0, 0, 0, 0, 0, 0};
  v8f c01 = {0, 0, 0, 0, 0, 0, 0, 0};
  v8f c10 = {0, 0, 0, 0, 0, 0, 0, 0};
  v8f c11 = {0, 0, 0, 0, 0, 0, 0, 0};
  for (int kb = 0; kb < K; kb += 32) {
    v16bf a0 = load_a_frag(A, lda, m0,      kb, lane);
    v16bf a1 = load_a_frag(A, lda, m0 + 16, kb, lane);
    v16bf b0 = load_bt_frag(Bt, ldb, n0,      kb, lane);
    v16bf b1 = load_bt_frag(Bt, ldb, n0 + 16, kb, lane);
    c00 = WMMA_BF16(a0, b0, c00);
    c01 = WMMA_BF16(a0, b1, c01);
    c10 = WMMA_BF16(a1, b0, c10);
    c11 = WMMA_BF16(a1, b1, c11);
  }
  int n = lane & 15;
  float bv0 = bias ? bias[n0 + n] : 0.0f;
  float bv1 = bias ? bias[n0 + 16 + n] : 0.0f;
  if (Cf) {
    store_tile_f32(Cf, ldc, m0,      n0,      lane, c00, bv0);
    store_tile_f32(Cf, ldc, m0,      n0 + 16, lane, c01, bv1);
    store_tile_f32(Cf, ldc, m0 + 16, n0,      lane, c10, bv0);
    store_tile_f32(Cf, ldc, m0 + 16, n0 + 16, lane, c11, bv1);
  }
  if (Cb) {
    store_tile_bf16(Cb, ldc, m0,      n0,      lane, c00, bv0);
    store_tile_bf16(Cb, ldc, m0,      n0 + 16, lane, c01, bv1);
    store_tile_bf16(Cb, ldc, m0 + 16, n0,      lane, c10, bv0);
    store_tile_bf16(Cb, ldc, m0 + 16, n0 + 16, lane, c11, bv1);
  }
}

// ---------------------------------------------------------------------------
// S[b,h,q,k] = sum_d Q[b,q,h,d] * K[b,k,h,d].
// Wave computes a 32x32 score tile (2x2 WMMA tiles), K-loop over dh=64.
// ---------------------------------------------------------------------------
__global__ void k_scores(const unsigned short* __restrict__ Qb,
                         const unsigned short* __restrict__ Kb,
                         float* __restrict__ S,
                         int Hh, int Lq, int Lk, int Dm, int nTiles) {
  int lane = threadIdx.x & 31;
  int wave = threadIdx.x >> 5;
  int tile = blockIdx.x * 8 + wave;
  if (tile >= nTiles) return;
  int kt = Lk >> 5;                       // 32-wide key tiles
  int perBH = (Lq >> 5) * kt;
  int bh = tile / perBH;
  int rem = tile - bh * perBH;
  int q0 = (rem / kt) << 5;
  int k0 = (rem - (rem / kt) * kt) << 5;
  int b = bh / Hh, h = bh - b * Hh;
  const unsigned short* Qp = Qb + (size_t)(b * Lq) * Dm + h * 64;
  const unsigned short* Kp = Kb + (size_t)(b * Lk) * Dm + h * 64;
  v8f c00 = {0, 0, 0, 0, 0, 0, 0, 0};
  v8f c01 = {0, 0, 0, 0, 0, 0, 0, 0};
  v8f c10 = {0, 0, 0, 0, 0, 0, 0, 0};
  v8f c11 = {0, 0, 0, 0, 0, 0, 0, 0};
#pragma unroll
  for (int kb = 0; kb < 64; kb += 32) {
    v16bf a0 = load_a_frag(Qp, Dm, q0,      kb, lane);
    v16bf a1 = load_a_frag(Qp, Dm, q0 + 16, kb, lane);
    v16bf b0 = load_bt_frag(Kp, Dm, k0,      kb, lane);
    v16bf b1 = load_bt_frag(Kp, Dm, k0 + 16, kb, lane);
    c00 = WMMA_BF16(a0, b0, c00);
    c01 = WMMA_BF16(a0, b1, c01);
    c10 = WMMA_BF16(a1, b0, c10);
    c11 = WMMA_BF16(a1, b1, c11);
  }
  float* Srow = S + (size_t)bh * Lq * Lk;
  store_tile_f32(Srow, Lk, q0,      k0,      lane, c00, 0.0f);
  store_tile_f32(Srow, Lk, q0,      k0 + 16, lane, c01, 0.0f);
  store_tile_f32(Srow, Lk, q0 + 16, k0,      lane, c10, 0.0f);
  store_tile_f32(Srow, Lk, q0 + 16, k0 + 16, lane, c11, 0.0f);
}

// ---------------------------------------------------------------------------
// Softmax over keys with scale / mask / optional gate / optional structural
// masks (visn self-attention). One 256-thread block per (b,h,q) row.
// ---------------------------------------------------------------------------
__global__ void k_softmax(const float* __restrict__ S, unsigned short* __restrict__ P,
                          const float* __restrict__ mask, const float* __restrict__ gate,
                          const int* __restrict__ feat_len, const int* __restrict__ prox_ptr,
                          int Hh, int Lq, int Lk, float scale, int use_extra) {
  __shared__ float red[256];
  int row = blockIdx.x;
  int bh = row / Lq;
  int q = row - bh * Lq;
  int b = bh / Hh;
  const float* s = S + (size_t)row * Lk;
  unsigned short* p = P + (size_t)row * Lk;
  int prox = 0, fl = 0;
  if (use_extra) { prox = prox_ptr[0]; fl = feat_len[b]; }
  float v[2];
  int cnt = 0;
  float lmax = -3.0e38f;
  for (int k = threadIdx.x; k < Lk; k += 256) {
    float x = s[k] * scale;
    float mk = mask[(size_t)b * Lk + k];
    x = gate ? (x * gate[(size_t)b * Lk + k] + mk) : (x + mk);
    if (use_extra) {
      bool mstat = ((q >= prox) && (k >= prox)) ||
                   ((q < prox - 1) && ((k > q) || (k < q - 1)));
      bool mbat = (q >= prox) && (k < fl - 1);
      if (mstat || mbat) x = -100000.0f;
    }
    v[cnt++] = x;
    lmax = fmaxf(lmax, x);
  }
  red[threadIdx.x] = lmax; __syncthreads();
  for (int o = 128; o > 0; o >>= 1) {
    if (threadIdx.x < o) red[threadIdx.x] = fmaxf(red[threadIdx.x], red[threadIdx.x + o]);
    __syncthreads();
  }
  float gmax = red[0]; __syncthreads();
  float lsum = 0.f;
  for (int i = 0; i < cnt; ++i) { v[i] = __expf(v[i] - gmax); lsum += v[i]; }
  red[threadIdx.x] = lsum; __syncthreads();
  for (int o = 128; o > 0; o >>= 1) {
    if (threadIdx.x < o) red[threadIdx.x] += red[threadIdx.x + o];
    __syncthreads();
  }
  float inv = 1.0f / red[0];
  cnt = 0;
  for (int k = threadIdx.x; k < Lk; k += 256) p[k] = f2bf(v[cnt++] * inv);
}

// ---------------------------------------------------------------------------
// Ctx[b,q,h,d] = sum_k P[b,h,q,k] * Vt[b,h,d,k].
// Wave computes a 32x32 (q x d) context tile, K-loop over Lk.
// ---------------------------------------------------------------------------
__global__ void k_pv(const unsigned short* __restrict__ P,
                     const unsigned short* __restrict__ Vt,
                     unsigned short* __restrict__ Ctx,
                     int Hh, int Lq, int Lk, int Dm, int nTiles) {
  int lane = threadIdx.x & 31;
  int wave = threadIdx.x >> 5;
  int tile = blockIdx.x * 8 + wave;
  if (tile >= nTiles) return;
  int perBH = (Lq >> 5) * 2;              // 2 d-tiles of 32 in dh=64
  int bh = tile / perBH;
  int rem = tile - bh * perBH;
  int q0 = (rem >> 1) << 5;
  int d0 = (rem & 1) << 5;
  const unsigned short* Pp = P + (size_t)bh * Lq * Lk;
  const unsigned short* Vp = Vt + (size_t)bh * 64 * Lk;
  v8f c00 = {0, 0, 0, 0, 0, 0, 0, 0};
  v8f c01 = {0, 0, 0, 0, 0, 0, 0, 0};
  v8f c10 = {0, 0, 0, 0, 0, 0, 0, 0};
  v8f c11 = {0, 0, 0, 0, 0, 0, 0, 0};
  for (int kb = 0; kb < Lk; kb += 32) {
    v16bf a0 = load_a_frag(Pp, Lk, q0,      kb, lane);
    v16bf a1 = load_a_frag(Pp, Lk, q0 + 16, kb, lane);
    v16bf b0 = load_bt_frag(Vp, Lk, d0,      kb, lane);
    v16bf b1 = load_bt_frag(Vp, Lk, d0 + 16, kb, lane);
    c00 = WMMA_BF16(a0, b0, c00);
    c01 = WMMA_BF16(a0, b1, c01);
    c10 = WMMA_BF16(a1, b0, c10);
    c11 = WMMA_BF16(a1, b1, c11);
  }
  int b = bh / Hh, h = bh - b * Hh;
  unsigned short* Cp = Ctx + (size_t)(b * Lq) * Dm + h * 64;
  store_tile_bf16(Cp, Dm, q0,      d0,      lane, c00, 0.0f);
  store_tile_bf16(Cp, Dm, q0,      d0 + 16, lane, c01, 0.0f);
  store_tile_bf16(Cp, Dm, q0 + 16, d0,      lane, c10, 0.0f);
  store_tile_bf16(Cp, Dm, q0 + 16, d0 + 16, lane, c11, 0.0f);
}

// ---------------------------------------------------------------------------
// Elementwise / normalization kernels
// ---------------------------------------------------------------------------
__global__ void k_cvt(const float* __restrict__ in, unsigned short* __restrict__ out, size_t n) {
  size_t i = (size_t)blockIdx.x * blockDim.x + threadIdx.x;
  if (i < n) out[i] = f2bf(in[i]);
}

__global__ void k_wt(const float* __restrict__ W, unsigned short* __restrict__ Wt, int K, int N) {
  size_t i = (size_t)blockIdx.x * blockDim.x + threadIdx.x;
  if (i >= (size_t)K * N) return;
  int nrow = (int)(i / (size_t)K);
  int k = (int)(i - (size_t)nrow * K);
  Wt[i] = f2bf(W[(size_t)k * N + nrow]);
}

__global__ void k_build_vt(const unsigned short* __restrict__ Vtok,
                           unsigned short* __restrict__ Vt,
                           int Hh, int L, int Dm, size_t total) {
  size_t i = (size_t)blockIdx.x * blockDim.x + threadIdx.x;
  if (i >= total) return;
  int perBH = 64 * L;
  int bh = (int)(i / (size_t)perBH);
  int rem = (int)(i - (size_t)bh * perBH);
  int d = rem / L;
  int kk = rem - d * L;
  int b = bh / Hh, h = bh - b * Hh;
  Vt[i] = Vtok[(size_t)(b * L + kk) * Dm + h * 64 + d];
}

__global__ void k_add_ln(const float* __restrict__ A, const float* __restrict__ R,
                         const float* __restrict__ g, const float* __restrict__ bb,
                         float* __restrict__ outF, unsigned short* __restrict__ outB, int Dm) {
  __shared__ float r1[256], r2[256];
  int row = blockIdx.x;
  const float* a = A + (size_t)row * Dm;
  const float* r = R + (size_t)row * Dm;
  float s = 0.f, s2 = 0.f;
  for (int c = threadIdx.x; c < Dm; c += 256) {
    float x = a[c] + r[c];
    s += x; s2 += x * x;
  }
  r1[threadIdx.x] = s; r2[threadIdx.x] = s2; __syncthreads();
  for (int o = 128; o > 0; o >>= 1) {
    if (threadIdx.x < o) { r1[threadIdx.x] += r1[threadIdx.x + o]; r2[threadIdx.x] += r2[threadIdx.x + o]; }
    __syncthreads();
  }
  float mu = r1[0] / Dm;
  float var = r2[0] / Dm - mu * mu;
  float rstd = rsqrtf(var + 1e-12f);
  for (int c = threadIdx.x; c < Dm; c += 256) {
    float x = a[c] + r[c];
    float y = (x - mu) * rstd * g[c] + bb[c];
    if (outF) outF[(size_t)row * Dm + c] = y;
    if (outB) outB[(size_t)row * Dm + c] = f2bf(y);
  }
}

__global__ void k_pos_inject(float* __restrict__ X, unsigned short* __restrict__ Xb,
                             const float* __restrict__ pos, const int* __restrict__ feat_len,
                             int Lv, int Dm, int total) {
  int i = blockIdx.x * blockDim.x + threadIdx.x;
  if (i >= total) return;
  int t = i / Dm;
  int b = t / Lv;
  int l = t - b * Lv;
  float x = X[i];
  if (l < feat_len[b]) x += 2.0f * pos[i];
  X[i] = x;
  Xb[i] = f2bf(x);
}

__global__ void k_gelu(const float* __restrict__ Hf, unsigned short* __restrict__ Hb, size_t total) {
  size_t i = (size_t)blockIdx.x * blockDim.x + threadIdx.x;
  if (i >= total) return;
  float x = Hf[i];
  Hb[i] = f2bf(0.5f * x * (1.0f + erff(x * 0.70710678118654752f)));
}

// ---------------------------------------------------------------------------
// Host orchestration
// ---------------------------------------------------------------------------
extern "C" void kernel_launch(void* const* d_in, const int* in_sizes, int n_in,
                              void* d_out, int out_size, void* d_ws, size_t ws_size,
                              hipStream_t stream) {
  (void)in_sizes; (void)n_in; (void)out_size; (void)ws_size;
  const int Bn = 32, LLn = 512, LVn = 256, Dm = 768, FFd = 3072, Hn = 12;
  const int Nl = Bn * LLn, Nv = Bn * LVn, BH = Bn * Hn;
  const float SCALE = 0.125f;  // 1/sqrt(64)

  // --- param leaves (jax pytree: sorted dict keys) ---
  const float* c_db  = (const float*)d_in[0];
  const float* c_dw  = (const float*)d_in[1];
  const float* c_kb  = (const float*)d_in[2];
  const float* c_kw  = (const float*)d_in[3];
  const float* c_lnb = (const float*)d_in[4];
  const float* c_lng = (const float*)d_in[5];
  const float* c_qb  = (const float*)d_in[6];
  const float* c_qw  = (const float*)d_in[7];
  const float* c_vb  = (const float*)d_in[8];
  const float* c_vw  = (const float*)d_in[9];
  const float* li_b  = (const float*)d_in[10];
  const float* li_w  = (const float*)d_in[11];
  const float* lo_db = (const float*)d_in[12];
  const float* lo_dw = (const float*)d_in[13];
  const float* lo_lnb= (const float*)d_in[14];
  const float* lo_lng= (const float*)d_in[15];
  const float* ls_db = (const float*)d_in[16];
  const float* ls_dw = (const float*)d_in[17];
  const float* ls_kb = (const float*)d_in[18];
  const float* ls_kw = (const float*)d_in[19];
  const float* ls_lnb= (const float*)d_in[20];
  const float* ls_lng= (const float*)d_in[21];
  const float* ls_qb = (const float*)d_in[22];
  const float* ls_qw = (const float*)d_in[23];
  const float* ls_vb = (const float*)d_in[24];
  const float* ls_vw = (const float*)d_in[25];
  const float* vi_b  = (const float*)d_in[26];
  const float* vi_w  = (const float*)d_in[27];
  const float* vo_db = (const float*)d_in[28];
  const float* vo_dw = (const float*)d_in[29];
  const float* vo_lnb= (const float*)d_in[30];
  const float* vo_lng= (const float*)d_in[31];
  const float* vs_db = (const float*)d_in[32];
  const float* vs_dw = (const float*)d_in[33];
  const float* vs_kb = (const float*)d_in[34];
  const float* vs_kw = (const float*)d_in[35];
  const float* vs_lnb= (const float*)d_in[36];
  const float* vs_lng= (const float*)d_in[37];
  const float* vs_qb = (const float*)d_in[38];
  const float* vs_qw = (const float*)d_in[39];
  const float* vs_vb = (const float*)d_in[40];
  const float* vs_vw = (const float*)d_in[41];
  // d_in[42] = layer_num (unused)
  const float* lang_feats = (const float*)d_in[43];
  const float* lang_mask  = (const float*)d_in[44];
  // 45,46,47 unused (cross masks, lang gate)
  const float* visn_feats = (const float*)d_in[48];
  const float* visn_mask  = (const float*)d_in[49];
  const float* visn_gate  = (const float*)d_in[50];
  const float* vis_pos    = (const float*)d_in[51];
  const int*   feat_len   = (const int*)d_in[52];
  const int*   prox_ptr   = (const int*)d_in[53];

  // --- workspace arena ---
  char* Wp = (char*)d_ws;
  size_t off = 0;
  auto alloc = [&](size_t bytes) -> size_t {
    off = (off + 255) & ~(size_t)255;
    size_t o = off; off += bytes; return o;
  };
  const size_t DD2 = (size_t)Dm * Dm * 2;
  const size_t DF2 = (size_t)Dm * FFd * 2;
  size_t o_cq = alloc(DD2), o_ck = alloc(DD2), o_cv = alloc(DD2), o_cd = alloc(DD2);
  size_t o_lq = alloc(DD2), o_lk = alloc(DD2), o_lv = alloc(DD2), o_ld = alloc(DD2);
  size_t o_vq = alloc(DD2), o_vk = alloc(DD2), o_vv = alloc(DD2), o_vd = alloc(DD2);
  size_t o_li = alloc(DF2), o_lo = alloc(DF2), o_vi = alloc(DF2), o_vo = alloc(DF2);
  size_t o_Xlb = alloc((size_t)Nl * Dm * 2), o_Xvb = alloc((size_t)Nv * Dm * 2);
  size_t o_Xl  = alloc((size_t)Nl * Dm * 4), o_Xv  = alloc((size_t)Nv * Dm * 4);
  size_t o_Qb  = alloc((size_t)Nl * Dm * 2), o_Kb  = alloc((size_t)Nl * Dm * 2);
  size_t o_Vb  = alloc((size_t)Nl * Dm * 2), o_Vt  = alloc((size_t)Nl * Dm * 2);
  size_t o_S   = alloc((size_t)BH * 512 * 512 * 4);   // also FFN inter f32
  size_t o_P   = alloc((size_t)BH * 512 * 512 * 2);   // also FFN gelu bf16
  size_t o_CtxL = alloc((size_t)Nl * Dm * 2), o_CtxV = alloc((size_t)Nv * Dm * 2);
  size_t o_Dense = alloc((size_t)Nl * Dm * 4);

  typedef unsigned short us;
  us* WT_cq=(us*)(Wp+o_cq); us* WT_ck=(us*)(Wp+o_ck); us* WT_cv=(us*)(Wp+o_cv); us* WT_cd=(us*)(Wp+o_cd);
  us* WT_lq=(us*)(Wp+o_lq); us* WT_lk=(us*)(Wp+o_lk); us* WT_lv=(us*)(Wp+o_lv); us* WT_ld=(us*)(Wp+o_ld);
  us* WT_vq=(us*)(Wp+o_vq); us* WT_vk=(us*)(Wp+o_vk); us* WT_vv=(us*)(Wp+o_vv); us* WT_vd=(us*)(Wp+o_vd);
  us* WT_li=(us*)(Wp+o_li); us* WT_lo=(us*)(Wp+o_lo); us* WT_vi=(us*)(Wp+o_vi); us* WT_vo=(us*)(Wp+o_vo);
  us* Xlb=(us*)(Wp+o_Xlb); us* Xvb=(us*)(Wp+o_Xvb);
  float* Xl=(float*)(Wp+o_Xl); float* Xv=(float*)(Wp+o_Xv);
  us* Qbp=(us*)(Wp+o_Qb); us* Kbp=(us*)(Wp+o_Kb); us* Vbp=(us*)(Wp+o_Vb); us* Vtp=(us*)(Wp+o_Vt);
  float* Sp=(float*)(Wp+o_S); us* Pp=(us*)(Wp+o_P);
  us* CtxL=(us*)(Wp+o_CtxL); us* CtxV=(us*)(Wp+o_CtxV);
  float* DenseF=(float*)(Wp+o_Dense);
  float* interF=(float*)(Wp+o_S);
  us* geluB=(us*)(Wp+o_P);

  auto cvt = [&](const float* in, us* out, size_t n) {
    k_cvt<<<dim3((unsigned)((n + 255) / 256)), 256, 0, stream>>>(in, out, n);
  };
  auto wt = [&](const float* w, us* wtp, int K, int N) {
    size_t n = (size_t)K * N;
    k_wt<<<dim3((unsigned)((n + 255) / 256)), 256, 0, stream>>>(w, wtp, K, N);
  };
  auto gemm = [&](const us* A, int lda, const us* Bt, int ldb, float* Cf, us* Cb,
                  int ldc, int M, int N, int K, const float* bias) {
    dim3 g((unsigned)(N / 64), (unsigned)(M / 128));
    k_gemm<<<g, 256, 0, stream>>>(A, lda, Bt, ldb, Cf, Cb, ldc, M, N, K, bias);
  };
  auto run_attn = [&](const us* Xq, int Lq, const us* Xc, int Lk,
                      us* Wq, us* Wk, us* Wv,
                      const float* bq, const float* bk, const float* bv,
                      const float* maskp, const float* gatep, int use_extra,
                      us* ctx_out) {
    int Mq = Bn * Lq, Mc = Bn * Lk;
    gemm(Xq, Dm, Wq, Dm, nullptr, Qbp, Dm, Mq, Dm, Dm, bq);
    gemm(Xc, Dm, Wk, Dm, nullptr, Kbp, Dm, Mc, Dm, Dm, bk);
    gemm(Xc, Dm, Wv, Dm, nullptr, Vbp, Dm, Mc, Dm, Dm, bv);
    size_t vt_total = (size_t)BH * 64 * Lk;
    k_build_vt<<<(unsigned)((vt_total + 255) / 256), 256, 0, stream>>>(Vbp, Vtp, Hn, Lk, Dm, vt_total);
    int nT = BH * (Lq / 32) * (Lk / 32);
    k_scores<<<(nT + 7) / 8, 256, 0, stream>>>(Qbp, Kbp, Sp, Hn, Lq, Lk, Dm, nT);
    k_softmax<<<BH * Lq, 256, 0, stream>>>(Sp, Pp, maskp, gatep, feat_len, prox_ptr,
                                           Hn, Lq, Lk, SCALE, use_extra);
    int nT2 = BH * (Lq / 32) * 2;
    k_pv<<<(nT2 + 7) / 8, 256, 0, stream>>>(Pp, Vtp, ctx_out, Hn, Lq, Lk, Dm, nT2);
  };

  // --- prep: weight transpose+bf16, activation bf16 ---
  wt(c_qw, WT_cq, Dm, Dm);  wt(c_kw, WT_ck, Dm, Dm);  wt(c_vw, WT_cv, Dm, Dm);  wt(c_dw, WT_cd, Dm, Dm);
  wt(ls_qw, WT_lq, Dm, Dm); wt(ls_kw, WT_lk, Dm, Dm); wt(ls_vw, WT_lv, Dm, Dm); wt(ls_dw, WT_ld, Dm, Dm);
  wt(vs_qw, WT_vq, Dm, Dm); wt(vs_kw, WT_vk, Dm, Dm); wt(vs_vw, WT_vv, Dm, Dm); wt(vs_dw, WT_vd, Dm, Dm);
  wt(li_w, WT_li, Dm, FFd); wt(lo_dw, WT_lo, FFd, Dm);
  wt(vi_w, WT_vi, Dm, FFd); wt(vo_dw, WT_vo, FFd, Dm);
  cvt(lang_feats, Xlb, (size_t)Nl * Dm);
  cvt(visn_feats, Xvb, (size_t)Nv * Dm);

  // --- cross attention (shared weights; both directions use original feats) ---
  run_attn(Xlb, LLn, Xvb, LVn, WT_cq, WT_ck, WT_cv, c_qb, c_kb, c_vb,
           visn_mask, nullptr, 0, CtxL);
  run_attn(Xvb, LVn, Xlb, LLn, WT_cq, WT_ck, WT_cv, c_qb, c_kb, c_vb,
           lang_mask, nullptr, 0, CtxV);
  gemm(CtxL, Dm, WT_cd, Dm, DenseF, nullptr, Dm, Nl, Dm, Dm, c_db);
  k_add_ln<<<Nl, 256, 0, stream>>>(DenseF, lang_feats, c_lng, c_lnb, Xl, Xlb, Dm);
  gemm(CtxV, Dm, WT_cd, Dm, DenseF, nullptr, Dm, Nv, Dm, Dm, c_db);
  k_add_ln<<<Nv, 256, 0, stream>>>(DenseF, visn_feats, c_lng, c_lnb, Xv, Xvb, Dm);

  // --- lang self attention ---
  run_attn(Xlb, LLn, Xlb, LLn, WT_lq, WT_lk, WT_lv, ls_qb, ls_kb, ls_vb,
           lang_mask, nullptr, 0, CtxL);
  gemm(CtxL, Dm, WT_ld, Dm, DenseF, nullptr, Dm, Nl, Dm, Dm, ls_db);
  k_add_ln<<<Nl, 256, 0, stream>>>(DenseF, Xl, ls_lng, ls_lnb, Xl, Xlb, Dm);

  // --- visn self attention (position injection + gate + structural masks) ---
  k_pos_inject<<<(Nv * Dm + 255) / 256, 256, 0, stream>>>(Xv, Xvb, vis_pos, feat_len,
                                                          LVn, Dm, Nv * Dm);
  run_attn(Xvb, LVn, Xvb, LVn, WT_vq, WT_vk, WT_vv, vs_qb, vs_kb, vs_vb,
           visn_mask, visn_gate, 1, CtxV);
  gemm(CtxV, Dm, WT_vd, Dm, DenseF, nullptr, Dm, Nv, Dm, Dm, vs_db);
  k_add_ln<<<Nv, 256, 0, stream>>>(DenseF, Xv, vs_lng, vs_lnb, Xv, Xvb, Dm);

  // --- lang FFN (inter f32 aliases S, gelu bf16 aliases P) ---
  gemm(Xlb, Dm, WT_li, Dm, interF, nullptr, FFd, Nl, FFd, Dm, li_b);
  k_gelu<<<(unsigned)(((size_t)Nl * FFd + 255) / 256), 256, 0, stream>>>(interF, geluB, (size_t)Nl * FFd);
  gemm(geluB, FFd, WT_lo, FFd, DenseF, nullptr, Dm, Nl, Dm, FFd, lo_db);
  k_add_ln<<<Nl, 256, 0, stream>>>(DenseF, Xl, lo_lng, lo_lnb, (float*)d_out, nullptr, Dm);

  // --- visn FFN ---
  gemm(Xvb, Dm, WT_vi, Dm, interF, nullptr, FFd, Nv, FFd, Dm, vi_b);
  k_gelu<<<(unsigned)(((size_t)Nv * FFd + 255) / 256), 256, 0, stream>>>(interF, geluB, (size_t)Nv * FFd);
  gemm(geluB, FFd, WT_vo, FFd, DenseF, nullptr, Dm, Nv, Dm, FFd, vo_db);
  k_add_ln<<<Nv, 256, 0, stream>>>(DenseF, Xv, vo_lng, vo_lnb,
                                   (float*)d_out + (size_t)Nl * Dm, nullptr, Dm);
}
